// NeuronMemory_28930899706062
// MI455X (gfx1250) — matrix operand order repaired and checked by reading, not compiled, not run on hardware
//
#include <hip/hip_runtime.h>
#include <hip/hip_bf16.h>
#include <math.h>

#define D_MODEL 1024
#define RANK 128
#define N_COMPRESS 16
#define N_KNOW 32768
#define K_TOP 8
#define NTOK 4096   // B*S = 2*2048

typedef __attribute__((ext_vector_type(16))) __bf16        v16bf;
typedef __attribute__((ext_vector_type(8)))  float         v8f;
typedef __attribute__((ext_vector_type(8)))  unsigned int  v8u;

union BF16Frag { v8u u; v16bf bf; uint4 q[2]; };

#if defined(__has_builtin)
#if __has_builtin(__builtin_amdgcn_tensor_load_to_lds) && __has_builtin(__builtin_amdgcn_s_wait_tensorcnt)
#define HAVE_TDM 1
#endif
#endif
#ifndef HAVE_TDM
#define HAVE_TDM 0
#endif

#if HAVE_TDM
typedef __attribute__((ext_vector_type(4))) unsigned int tdm_g0_t;
typedef __attribute__((ext_vector_type(8))) int          tdm_g1_t;
typedef __attribute__((ext_vector_type(4))) int          tdm_g23_t;

// Issue a TDM load of a [16 rows x 128 cols] bf16 tile of knowledge_K into LDS.
// D# per CDNA5 ISA ch.8: 2D tensor, data_size=2B, tile_dim0=128, tile_dim1=16,
// tensor_dim0_stride=128; groups 2/3 zero (<=2D). 6-arg clang-23 builtin form.
__device__ __forceinline__ void tdm_load_ktile(const unsigned short* gptr,
                                               unsigned lds_off) {
    unsigned long long ga = (unsigned long long)(uintptr_t)gptr;
    tdm_g0_t g0 = { 1u,                                   // count=1, user desc
                    lds_off,                              // lds_addr [63:32]
                    (unsigned)ga,                         // global_addr[31:0]
                    (unsigned)((ga >> 32) & 0x01FFFFFFull) | 0x80000000u }; // ga[56:32] | type=2
    tdm_g1_t g1 = { 0x00010000,        // workgroup_mask=0, data_size=1 (2 bytes)
                    0x00800000,        // tensor_dim0 = 128 (bits 63:48)
                    (int)0x80000000u,  // tensor_dim1 = 32768 (low16 at bits 95:80)
                    0x00800000,        // tile_dim0 = 128 (bits 127:112)
                    16,                // tile_dim1 = 16
                    128,               // tensor_dim0_stride = 128 (bits 191:160)
                    0, 0 };
    tdm_g23_t gz = { 0, 0, 0, 0 };
    tdm_g1_t  gz8 = { 0, 0, 0, 0, 0, 0, 0, 0 };
    __builtin_amdgcn_tensor_load_to_lds(g0, g1, gz, gz, gz8, 0);
}
#endif

__device__ __forceinline__ unsigned short f32_to_bf16(float f) {
    unsigned int u = __float_as_uint(f);
    unsigned int r = u + 0x7FFFu + ((u >> 16) & 1u);
    return (unsigned short)(r >> 16);
}

// ---------------- conversion kernels ----------------
__global__ void k_conv_bf16(const float* __restrict__ src,
                            unsigned short* __restrict__ dst, int n) {
    int i = blockIdx.x * blockDim.x + threadIdx.x;
    if (i < n) dst[i] = f32_to_bf16(src[i]);
}

// compress_neurons [16][1024][128] -> W2T[c][d] bf16 with c = n*128 + r
__global__ void k_conv_w2t(const float* __restrict__ cn,
                           unsigned short* __restrict__ w2t) {
    int i = blockIdx.x * blockDim.x + threadIdx.x; // over 2048*1024
    int d = i & (D_MODEL - 1);
    int c = i >> 10;
    int n = c >> 7;
    int r = c & (RANK - 1);
    w2t[i] = f32_to_bf16(cn[(size_t)n * (D_MODEL * RANK) + (size_t)d * RANK + r]);
}

// ---------------- router + softmax ----------------
// one wave per token
__global__ void __launch_bounds__(32)
k_router(const float* __restrict__ x, const float* __restrict__ rw,
         float* __restrict__ wts) {
    const int t    = blockIdx.x;
    const int lane = threadIdx.x;
    const int n    = lane & 15;
    const int half = lane >> 4;
    const float* xr = x  + (size_t)t * D_MODEL + half * 512;
    const float* wr = rw + (size_t)n * D_MODEL + half * 512;
    float acc = 0.f;
    for (int d = 0; d < 512; ++d) acc += xr[d] * wr[d];
    acc += __shfl_xor(acc, 16, 32);          // combine halves; lanes 0-15 valid
    float m = acc;
    for (int off = 8; off >= 1; off >>= 1) m = fmaxf(m, __shfl_xor(m, off, 16));
    float e = __expf(acc - m);
    float s = e;
    for (int off = 8; off >= 1; off >>= 1) s += __shfl_xor(s, off, 16);
    if (lane < 16) wts[(size_t)t * N_COMPRESS + n] = e / s;
}

// ---------------- compressor: Q = sum_n w_n * (x @ W_n) ----------------
// grid (256 token tiles, 8 r tiles), one wave each
__global__ void __launch_bounds__(32)
k_compress(const unsigned short* __restrict__ xb,   // [4096][1024] bf16
           const unsigned short* __restrict__ w2t,  // [2048][1024] bf16
           const float* __restrict__ wts,           // [4096][16]
           unsigned short* __restrict__ qb)         // [4096][128] bf16, pre-scaled
{
    __shared__ unsigned short lds_x[16 * D_MODEL]; // 32 KB
    __shared__ float lds_w[16 * 16];
    const int lane = threadIdx.x;
    const int t0   = blockIdx.x * 16;
    const int rt   = blockIdx.y;
    // stage x tile (16 rows x 1024 bf16) into LDS
    {
        const uint4* src = (const uint4*)(xb + (size_t)t0 * D_MODEL);
        uint4* dst = (uint4*)lds_x;
        for (int i = lane; i < (16 * D_MODEL) / 8; i += 32) dst[i] = src[i];
    }
    for (int i = lane; i < 256; i += 32) lds_w[i] = wts[(size_t)t0 * 16 + i];
    __syncthreads();

    const int m  = lane & 15;
    const int hi = lane >> 4;
    const unsigned short* arow = lds_x + m * D_MODEL + hi * 16;
    v8f acc = {};
    for (int n = 0; n < N_COMPRESS; ++n) {
        const unsigned short* bcol =
            w2t + ((size_t)(n * RANK + rt * 16 + m)) * D_MODEL + hi * 16;
        v8f c = {};
        for (int kc = 0; kc < D_MODEL / 32; ++kc) {
            BF16Frag A, Bm;
            A.q[0]  = *(const uint4*)(arow + kc * 32);
            A.q[1]  = *(const uint4*)(arow + kc * 32 + 8);
            Bm.q[0] = *(const uint4*)(bcol + kc * 32);
            Bm.q[1] = *(const uint4*)(bcol + kc * 32 + 8);
            c = __builtin_amdgcn_wmma_f32_16x16x32_bf16(
                    false, A.bf, false, Bm.bf, (short)0, c, false, false);
        }
#pragma unroll
        for (int v = 0; v < 8; ++v)
            acc[v] += lds_w[(v + hi * 8) * 16 + n] * c[v];
    }
    const float scale = 0.08838834764831845f; // 1/sqrt(128) folded into Q
#pragma unroll
    for (int v = 0; v < 8; ++v)
        qb[(size_t)(t0 + v + hi * 8) * RANK + rt * 16 + m] =
            f32_to_bf16(acc[v] * scale);
}

// ---------------- knowledge scores + fused top-8 + softmax ----------------
// one wave per 16-token tile; double-buffered TDM streams K tiles into LDS
__global__ void __launch_bounds__(32)
k_scores_topk(const unsigned short* __restrict__ qb,  // [4096][128] bf16
              const unsigned short* __restrict__ kb,  // [32768][128] bf16
              float* __restrict__ topw,               // [4096][8] softmax weights
              int*   __restrict__ topi)               // [4096][8]
{
    // NOTE: lds_kb declared first -> LDS offset 0 (TDM descriptor assumes this)
    __shared__ unsigned short lds_kb[2 * 16 * RANK];  // 2 x 4KB staging
    __shared__ float lds_s[256];
    const int lane = threadIdx.x;
    const int t0   = blockIdx.x * 16;
    const int m    = lane & 15;
    const int hi   = lane >> 4;

    BF16Frag A[4];
    {
        const unsigned short* arow = qb + (size_t)(t0 + m) * RANK + hi * 16;
#pragma unroll
        for (int c4 = 0; c4 < 4; ++c4) {
            A[c4].q[0] = *(const uint4*)(arow + c4 * 32);
            A[c4].q[1] = *(const uint4*)(arow + c4 * 32 + 8);
        }
    }

    float tv[K_TOP]; int ti[K_TOP];
#pragma unroll
    for (int q = 0; q < K_TOP; ++q) { tv[q] = -3.0e38f; ti[q] = 0; }
    float tmin = -3.0e38f; int minpos = 0;

    const int NT = N_KNOW / 16;   // 2048 tiles
#if HAVE_TDM
    tdm_load_ktile(kb, 0);        // prime buffer 0
#endif
    for (int nt = 0; nt < NT; ++nt) {
        v8f c = {};
#if HAVE_TDM
        if (nt + 1 < NT) {
            tdm_load_ktile(kb + (size_t)(nt + 1) * 16 * RANK,
                           (unsigned)(((nt + 1) & 1) * 16 * RANK * 2));
            __builtin_amdgcn_s_wait_tensorcnt(1);  // tile nt complete (in-order TDM)
        } else {
            __builtin_amdgcn_s_wait_tensorcnt(0);
        }
        asm volatile("" ::: "memory");
        const unsigned short* bcol =
            lds_kb + (nt & 1) * (16 * RANK) + m * RANK + hi * 16;
#else
        const unsigned short* bcol =
            kb + (size_t)(nt * 16 + m) * RANK + hi * 16;
        __builtin_prefetch(bcol + 16 * RANK, 0, 0);
#endif
#pragma unroll
        for (int c4 = 0; c4 < 4; ++c4) {
            BF16Frag Bm;
            Bm.q[0] = *(const uint4*)(bcol + c4 * 32);
            Bm.q[1] = *(const uint4*)(bcol + c4 * 32 + 8);
            c = __builtin_amdgcn_wmma_f32_16x16x32_bf16(
                    false, A[c4].bf, false, Bm.bf, (short)0, c, false, false);
        }
#pragma unroll
        for (int v = 0; v < 8; ++v) lds_s[(v + hi * 8) * 16 + m] = c[v];
        __syncthreads();
        if (lane < 16) {
#pragma unroll
            for (int j = 0; j < 16; ++j) {
                float v = lds_s[lane * 16 + j];
                if (v > tmin) {
                    int idx = nt * 16 + j;
#pragma unroll
                    for (int q = 0; q < K_TOP; ++q)
                        if (q == minpos) { tv[q] = v; ti[q] = idx; }
                    tmin = tv[0]; minpos = 0;
#pragma unroll
                    for (int q = 1; q < K_TOP; ++q)
                        if (tv[q] < tmin) { tmin = tv[q]; minpos = q; }
                }
            }
        }
        __syncthreads();
    }
    if (lane < 16) {
        float mx = tv[0];
#pragma unroll
        for (int q = 1; q < K_TOP; ++q) mx = fmaxf(mx, tv[q]);
        float e[K_TOP]; float s = 0.f;
#pragma unroll
        for (int q = 0; q < K_TOP; ++q) { e[q] = __expf(tv[q] - mx); s += e[q]; }
        float inv = 1.0f / s;
        int t = t0 + lane;
#pragma unroll
        for (int q = 0; q < K_TOP; ++q) {
            topw[(size_t)t * K_TOP + q] = e[q] * inv;
            topi[(size_t)t * K_TOP + q] = ti[q];
        }
    }
}

// ---------------- gather V + weighted sum ----------------
__global__ void __launch_bounds__(256)
k_gather(const float* __restrict__ V, const float* __restrict__ topw,
         const int* __restrict__ topi, float* __restrict__ out)
{
    const int t   = blockIdx.x;
    const int tid = threadIdx.x;
    float w[K_TOP]; const float* vp[K_TOP];
#pragma unroll
    for (int q = 0; q < K_TOP; ++q) {
        w[q]  = topw[(size_t)t * K_TOP + q];
        vp[q] = V + (size_t)topi[(size_t)t * K_TOP + q] * D_MODEL;
        __builtin_prefetch(vp[q] + tid, 0, 0);
    }
    float* op = out + (size_t)t * D_MODEL;
    for (int c = tid; c < D_MODEL; c += 256) {
        float acc = 0.f;
#pragma unroll
        for (int q = 0; q < K_TOP; ++q) acc += w[q] * vp[q][c];
        op[c] = acc;
    }
}

// ---------------- launch ----------------
extern "C" void kernel_launch(void* const* d_in, const int* in_sizes, int n_in,
                              void* d_out, int out_size, void* d_ws, size_t ws_size,
                              hipStream_t stream) {
    const float* x  = (const float*)d_in[0];
    const float* rw = (const float*)d_in[1];
    const float* cn = (const float*)d_in[2];
    const float* kK = (const float*)d_in[3];
    const float* kV = (const float*)d_in[4];
    float* out = (float*)d_out;

    char* ws = (char*)d_ws;
    unsigned short* xb  = (unsigned short*)(ws);                       // 8,388,608 B
    unsigned short* w2t = (unsigned short*)(ws + 8388608);             // 4,194,304 B
    unsigned short* kb  = (unsigned short*)(ws + 12582912);            // 8,388,608 B
    float*          wts = (float*)        (ws + 20971520);             //   262,144 B
    unsigned short* qb  = (unsigned short*)(ws + 21233664);            // 1,048,576 B
    float*          topw= (float*)        (ws + 22282240);             //   131,072 B
    int*            topi= (int*)          (ws + 22413312);             //   131,072 B

    const int nx = NTOK * D_MODEL;            // 4,194,304
    const int nk = N_KNOW * RANK;             // 4,194,304
    k_conv_bf16<<<nx / 256, 256, 0, stream>>>(x, xb, nx);
    k_conv_w2t<<<(N_COMPRESS * RANK * D_MODEL) / 256, 256, 0, stream>>>(cn, w2t);
    k_conv_bf16<<<nk / 256, 256, 0, stream>>>(kK, kb, nk);
    k_router<<<NTOK, 32, 0, stream>>>(x, rw, wts);
    k_compress<<<dim3(NTOK / 16, RANK / 16), 32, 0, stream>>>(xb, w2t, wts, qb);
    k_scores_topk<<<NTOK / 16, 32, 0, stream>>>(qb, kb, topw, topi);
    k_gather<<<NTOK, 256, 0, stream>>>(kV, topw, topi, out);
}
